// CnnRnn_28003186770662
// MI455X (gfx1250) — matrix operand order repaired and checked by reading, not compiled, not run on hardware
//
#include <hip/hip_runtime.h>
#include <hip/hip_bf16.h>

typedef __attribute__((ext_vector_type(16))) _Float16 v16h;
typedef __attribute__((ext_vector_type(8)))  _Float16 v8h;
typedef __attribute__((ext_vector_type(8)))  float    v8f;
typedef __attribute__((ext_vector_type(4)))  float    f4;

#define Hdim 256
#define Wdim 256
#define Ddim 512
#define HID  128
#define SEQN 1024   // 32*32

// ---------------------------------------------------------------------------
// 1) depth_map passthrough: 268 MB streaming, float4-vectorized, non-temporal
//    so the dead stream does not displace the turning kernel's L2 working set.
// ---------------------------------------------------------------------------
__global__ __launch_bounds__(256) void k_copy_depth(const f4* __restrict__ src,
                                                    f4* __restrict__ dst, int n4) {
    int i = blockIdx.x * 256 + threadIdx.x;
    if (i < n4) {
        f4 v = __builtin_nontemporal_load(src + i);
        __builtin_nontemporal_store(v, dst + i);
    }
}

// ---------------------------------------------------------------------------
// 2) Pack Wx (1024x128 f32, row-major K x N) into this GEMM's contiguous-k WMMA B-operand
//    per-lane layout, f16.  Tile (kt 0..31, tn 0..7), lane l:
//      col = tn*16 + (l&15),  K = kt*32 + (l>>4)*16 + jj  (jj = 0..15)
//    Stored so the GEMM reads one contiguous 32B v16h per (tile, lane).
// ---------------------------------------------------------------------------
__global__ __launch_bounds__(256) void k_pack_wxb(const float* __restrict__ Wx,
                                                  _Float16* __restrict__ WxB) {
    int tid = blockIdx.x * 256 + threadIdx.x;   // 0..8191 (256 tiles * 32 lanes)
    int l  = tid & 31;
    int tile = tid >> 5;                        // kt*8 + tn
    int kt = tile >> 3;
    int tn = tile & 7;
    int lr = l & 15;
    int kg = l >> 4;
    int kb = kt * 32 + kg * 16;
    int nc = tn * 16 + lr;
    _Float16* dst = WxB + (size_t)tid * 16;
    #pragma unroll
    for (int jj = 0; jj < 16; ++jj)
        dst[jj] = (_Float16)Wx[(size_t)(kb + jj) * HID + nc];
}

// ---------------------------------------------------------------------------
// 3) turning(): f[i,j,d] = mean over 4 anchors x 3x3 taps of
//    u[X*32/15, Y*32/15, d],  u = feature * (depth >= 0.1)
//    Written directly as f16 seq[d][i*32+j] for the WMMA GEMM.
//    Lane dim = d -> anchor reads coalesced; gathers are L2-resident (~59MB).
// ---------------------------------------------------------------------------
__global__ __launch_bounds__(256) void k_turning(const float* __restrict__ depth,
                                                 const float* __restrict__ feat,
                                                 const int*   __restrict__ anchors,
                                                 _Float16*    __restrict__ seq16) {
    int tid = blockIdx.x * 256 + threadIdx.x;   // 0 .. 512*1024-1
    int d = tid & (Ddim - 1);
    int n = tid >> 9;                           // 0..1023
    int i = n >> 5, j = n & 31;
    const int* a0 = anchors;                    // [4096][512]
    const int* a1 = anchors + 4096 * Ddim;
    float s = 0.f;
    #pragma unroll
    for (int da1 = 0; da1 < 2; ++da1) {
        #pragma unroll
        for (int da2 = 0; da2 < 2; ++da2) {
            int a  = (2 * j + da1) * 64 + (2 * i + da2);   // a = a1*64 + a2
            int A0 = a0[(size_t)a * Ddim + d];
            int A1 = a1[(size_t)a * Ddim + d];
            #pragma unroll
            for (int p = 0; p < 3; ++p) {
                unsigned X  = (unsigned)(A1 + p - 1);
                unsigned gx = (X * 32u) / 15u;             // == X*256//120
                #pragma unroll
                for (int q = 0; q < 3; ++q) {
                    unsigned Y  = (unsigned)(A0 + q - 1);
                    unsigned gy = (Y * 32u) / 15u;
                    size_t off = ((size_t)gx * Wdim + gy) * Ddim + d;
                    float dm = depth[off];
                    float fv = feat[off];
                    s += (dm >= 0.1f) ? fv : 0.f;
                }
            }
        }
    }
    seq16[(size_t)d * SEQN + n] = (_Float16)(s * (1.f / 36.f));
}

// ---------------------------------------------------------------------------
// 4) XW = seq(512x1024) @ Wx(1024x128) + b   via v_wmma_f32_16x16x32_f16
//    32 blocks (M-tiles) x 8 waves (N-tiles).  Per K-step per wave:
//    A = two contiguous v8h loads, B = one contiguous v16h load (pre-packed),
//    one WMMA.  No scalar-load staircase.
// ---------------------------------------------------------------------------
__global__ __launch_bounds__(256) void k_gemm_xw(const _Float16* __restrict__ X,    // [512][1024]
                                                 const _Float16* __restrict__ WxB,  // packed B tiles
                                                 const float*    __restrict__ bias, // [128]
                                                 float*          __restrict__ XW) { // [512][128]
    int tm = blockIdx.x;            // 0..31  (M tile)
    int tn = threadIdx.x >> 5;      // 0..7   (N tile = wave id)
    int l  = threadIdx.x & 31;
    int lr = l & 15;
    int kg = l >> 4;

    v8f c;
    #pragma unroll
    for (int r = 0; r < 8; ++r) c[r] = 0.f;

    int trow = tm * 16 + lr;
    const _Float16* xrow = X + (size_t)trow * SEQN;
    for (int kt = 0; kt < 32; ++kt) {
        // A 16x32 f16: lane l holds row (l&15), K = kg*16..+15
        int kbase = kt * 32 + kg * 16;
        union { v16h v; v8h h[2]; } A;
        A.h[0] = *(const v8h*)(xrow + kbase);
        A.h[1] = *(const v8h*)(xrow + kbase + 8);
        // B: pre-packed per-lane 32B chunk
        v16h Bv = *(const v16h*)(WxB + ((size_t)(kt * 8 + tn) * 32 + l) * 16);
        c = __builtin_amdgcn_wmma_f32_16x16x32_f16(false, A.v, false, Bv,
                                                   (short)0, c, false, false);
    }
    float bn = bias[tn * 16 + lr];
    #pragma unroll
    for (int r = 0; r < 8; ++r) {
        int M = r + 8 * kg;                     // C/D layout
        XW[(size_t)(tm * 16 + M) * HID + tn * 16 + lr] = c[r] + bn;
    }
}

// ---------------------------------------------------------------------------
// 5) Recurrence: h_t = tanh(XW[t] + h_{t-1} @ Wh);  rppg[t] = h_t . Wo
//    One workgroup, 8 waves. Wh held in per-wave WMMA B-tiles (registers).
//    Only row 0 of A carries h (rows 1..15 zero) -> row 0 of D is the matvec.
// ---------------------------------------------------------------------------
__global__ __launch_bounds__(256) void k_rnn(const float* __restrict__ XW,   // [512][128]
                                             const float* __restrict__ Wh,   // [128][128]
                                             const float* __restrict__ Wo,   // [128]
                                             float* __restrict__ rppg) {     // [512]
    __shared__ _Float16 h16[HID];
    __shared__ float    h32[HID];
    int tid = threadIdx.x;
    int w   = tid >> 5;        // wave 0..7  -> N columns w*16..w*16+15
    int l   = tid & 31;
    int lr  = l & 15;
    int kg  = l >> 4;
    int ncol = w * 16 + lr;

    // Preload Wh as 4 B-tiles (K = kt*32..+31) per wave, f32 -> f16 (one-time)
    union { v16h v; _Float16 e[16]; } B[4];
    #pragma unroll
    for (int kt = 0; kt < 4; ++kt) {
        int kb = kt * 32 + kg * 16;
        #pragma unroll
        for (int jj = 0; jj < 16; ++jj)
            B[kt].e[jj] = (_Float16)Wh[(size_t)(kb + jj) * HID + ncol];
    }
    // Wo: 4 coeffs per lane for the wave-0 reduction
    float wo0 = Wo[l * 4 + 0], wo1 = Wo[l * 4 + 1];
    float wo2 = Wo[l * 4 + 2], wo3 = Wo[l * 4 + 3];

    if (tid < HID) { h16[tid] = (_Float16)0.f; h32[tid] = 0.f; }
    __syncthreads();

    for (int t = 0; t < 512; ++t) {
        v8f c;
        #pragma unroll
        for (int r = 0; r < 8; ++r) c[r] = 0.f;
        #pragma unroll
        for (int kt = 0; kt < 4; ++kt) {
            union { v16h v; _Float16 e[16]; } A;
            if (lr == 0) {                      // row 0 of A = h chunk
                int base = kt * 32 + kg * 16;
                #pragma unroll
                for (int ii = 0; ii < 8; ++ii) {
                    A.e[ii]     = h16[base + ii];
                    A.e[8 + ii] = h16[base + 8 + ii];
                }
            } else {
                #pragma unroll
                for (int ii = 0; ii < 16; ++ii) A.e[ii] = (_Float16)0.f;
            }
            c = __builtin_amdgcn_wmma_f32_16x16x32_f16(false, A.v, false, B[kt].v,
                                                       (short)0, c, false, false);
        }
        float hn = 0.f;
        if (kg == 0)                            // lanes 0..15 hold M=0 row in c[0]
            hn = tanhf(c[0] + XW[(size_t)t * HID + ncol]);
        __syncthreads();                        // all reads of h_{t-1} done
        if (kg == 0) { h32[ncol] = hn; h16[ncol] = (_Float16)hn; }
        __syncthreads();                        // h_t visible
        if (w == 0) {
            float s = h32[l * 4 + 0] * wo0 + h32[l * 4 + 1] * wo1 +
                      h32[l * 4 + 2] * wo2 + h32[l * 4 + 3] * wo3;
            #pragma unroll
            for (int off = 16; off > 0; off >>= 1) s += __shfl_down(s, off, 32);
            if (l == 0) rppg[t] = s;
        }
    }
}

// ---------------------------------------------------------------------------
extern "C" void kernel_launch(void* const* d_in, const int* in_sizes, int n_in,
                              void* d_out, int out_size, void* d_ws, size_t ws_size,
                              hipStream_t stream) {
    const float* depth = (const float*)d_in[0];
    const float* feat  = (const float*)d_in[1];
    const float* Wx    = (const float*)d_in[2];
    const float* Wh    = (const float*)d_in[3];
    const float* b     = (const float*)d_in[4];
    const float* Wo    = (const float*)d_in[5];
    const int* anchors = (const int*)d_in[6];
    // d_in[7] = turned (always 1 in reference setup)

    float* out = (float*)d_out;
    const size_t NDEPTH = (size_t)Hdim * Wdim * Ddim;   // 33,554,432
    float* out_rppg = out + NDEPTH;

    char* ws = (char*)d_ws;
    _Float16* seq16 = (_Float16*)ws;                                  // 1 MiB
    _Float16* WxB   = (_Float16*)(ws + (1u << 20));                   // 256 KiB (packed)
    float*    XW    = (float*)(ws + (1u << 20) + (256u << 10));       // 256 KiB

    int n4 = (int)(NDEPTH / 4);
    k_copy_depth<<<(n4 + 255) / 256, 256, 0, stream>>>((const f4*)depth,
                                                       (f4*)out, n4);
    k_pack_wxb<<<32, 256, 0, stream>>>(Wx, WxB);
    k_turning<<<(Ddim * SEQN) / 256, 256, 0, stream>>>(depth, feat, anchors, seq16);
    k_gemm_xw<<<32, 256, 0, stream>>>(seq16, WxB, b, XW);
    k_rnn<<<1, 256, 0, stream>>>(XW, Wh, Wo, out_rppg);
}
